// SPADEResnetBlock_1803886264394
// MI455X (gfx1250) — matrix operand order repaired and checked by reading, not compiled, not run on hardware
//
#include <hip/hip_runtime.h>
#include <hip/hip_bf16.h>
#include <stdint.h>

// ---------------------------------------------------------------------------
// SPADE ResNet block for MI455X (gfx1250), bf16 WMMA implicit-GEMM conv3d.
// B=4, S=32^3, FIN=64, FMID=FOUT=32, NH=64, NCLS=4.
// Activations kept channels-last [b][pos][ci] in bf16 so that:
//   - LDS halo staging is a contiguous b128 copy (async global->LDS on CDNA5)
//   - WMMA B fragments are contiguous 32B LDS reads
//   - epilogue stores are contiguous 32B per lane
// ---------------------------------------------------------------------------

typedef __attribute__((ext_vector_type(16))) __bf16 v16bf;
typedef __attribute__((ext_vector_type(8)))  __bf16 v8bf;
typedef __attribute__((ext_vector_type(8)))  float  v8f;
typedef __attribute__((ext_vector_type(4)))  float  v4f;

#define SDIM  32
#define NSPAT (SDIM * SDIM * SDIM)   // 32768
#define NB    4

// CDNA5 async global->LDS staging (bypasses VGPRs, tracked by ASYNCcnt)
#if defined(__has_builtin)
#if __has_builtin(__builtin_amdgcn_global_load_async_to_lds_b128)
#define USE_ASYNC_LDS 1
#endif
#endif

#if defined(USE_ASYNC_LDS)
typedef __attribute__((__vector_size__(16))) int i32x4;
typedef __attribute__((address_space(1))) i32x4 as1_i32x4;
typedef __attribute__((address_space(3))) i32x4 as3_i32x4;
__device__ __forceinline__ void wait_asynccnt0() {
#if __has_builtin(__builtin_amdgcn_s_wait_asynccnt)
  __builtin_amdgcn_s_wait_asynccnt(0);
#else
  asm volatile("s_wait_asynccnt 0" ::: "memory");
#endif
}
#endif

__device__ __forceinline__ unsigned short f2bf(float f) {
  unsigned u = __float_as_uint(f);
  unsigned r = u + 0x7FFFu + ((u >> 16) & 1u);
  return (unsigned short)(r >> 16);
}

// --- weight transpose + convert: f32 [NC][Cout][Cin][T] -> bf16 [NC][T][Cout][Cin]
__global__ void wtrans_kernel(const float* __restrict__ w,
                              unsigned short* __restrict__ o,
                              int Cout, int Cin, int T, int total) {
  int i = blockIdx.x * blockDim.x + threadIdx.x;
  if (i >= total) return;
  int t  = i % T;
  int r  = i / T;
  int ci = r % Cin;  r /= Cin;
  int co = r % Cout;
  int n  = r / Cout;
  o[(((size_t)n * T + t) * Cout + co) * Cin + ci] = f2bf(w[i]);
}

// --- NCDHW f32 -> channels-last bf16, one (b,z,y) row per block, LDS transpose
template<int C>
__global__ __launch_bounds__(256)
void cvt_cl_kernel(const float* __restrict__ x, unsigned short* __restrict__ out) {
  __shared__ unsigned short sh[C * SDIM];
  const int row = blockIdx.x;            // b*1024 + z*32 + y
  const int b   = row >> 10;
  const int rp  = row & 1023;
  const size_t rowbase = (size_t)b * NSPAT + (size_t)rp * SDIM;
  for (int e = threadIdx.x; e < C * SDIM; e += 256) {
    const int ci = e >> 5, xc = e & 31;                       // [ci][x] -> coalesced read
    sh[xc * C + ci] = f2bf(x[((size_t)(b * C + ci)) * NSPAT + (size_t)rp * SDIM + xc]);
  }
  __syncthreads();
  for (int e = threadIdx.x; e < C * SDIM; e += 256)           // coalesced write
    out[rowbase * C + e] = sh[e];
}

// --- per-(b,c) mean/rstd, NCDHW layout (for input x)
__global__ __launch_bounds__(256)
void stats_kernel(const float* __restrict__ x,
                  float* __restrict__ mean, float* __restrict__ rstd) {
  int bc = blockIdx.x;
  const float* p = x + (size_t)bc * NSPAT;
  float s = 0.f, s2 = 0.f;
  for (int i = threadIdx.x; i < NSPAT; i += 256) { float v = p[i]; s += v; s2 += v * v; }
  __shared__ float sh[512];
  sh[threadIdx.x] = s; sh[256 + threadIdx.x] = s2;
  __syncthreads();
  for (int o = 128; o > 0; o >>= 1) {
    if (threadIdx.x < o) {
      sh[threadIdx.x] += sh[threadIdx.x + o];
      sh[256 + threadIdx.x] += sh[256 + threadIdx.x + o];
    }
    __syncthreads();
  }
  if (threadIdx.x == 0) {
    float m = sh[0] * (1.f / NSPAT);
    float var = sh[256] * (1.f / NSPAT) - m * m;
    mean[bc] = m; rstd[bc] = rsqrtf(var + 1e-5f);
  }
}

// --- per-(b,c) mean/rstd, channels-last layout (for dx); one block per b
template<int C>
__global__ __launch_bounds__(256)
void stats_cl_kernel(const float* __restrict__ x,
                     float* __restrict__ mean, float* __restrict__ rstd) {
  constexpr int G = 256 / C;
  const int b = blockIdx.x;
  const int c = threadIdx.x & (C - 1);
  const int grp = threadIdx.x / C;
  const float* p = x + (size_t)b * NSPAT * C;
  float s = 0.f, s2 = 0.f;
  for (int pos = grp; pos < NSPAT; pos += G) {           // fully coalesced
    float v = p[(size_t)pos * C + c];
    s += v; s2 += v * v;
  }
  __shared__ float sh[512];
  sh[threadIdx.x] = s; sh[256 + threadIdx.x] = s2;
  __syncthreads();
  for (int o = 128; o >= C; o >>= 1) {
    if (threadIdx.x < o) {
      sh[threadIdx.x] += sh[threadIdx.x + o];
      sh[256 + threadIdx.x] += sh[256 + threadIdx.x + o];
    }
    __syncthreads();
  }
  if (threadIdx.x < C) {
    float m = sh[threadIdx.x] * (1.f / NSPAT);
    float var = sh[256 + threadIdx.x] * (1.f / NSPAT) - m * m;
    mean[b * C + threadIdx.x] = m;
    rstd[b * C + threadIdx.x] = rsqrtf(var + 1e-5f);
  }
}

// --- SPADE combine: out_bf16_CL = lrelu( norm(x) * (1+gamma) + beta )
//     gamma/beta/out channels-last; x is NCDHW (XCL=false) or CL (XCL=true)
template<int C, bool XCL>
__global__ __launch_bounds__(256)
void spade_combine_kernel(const float* __restrict__ x,
                          const float* __restrict__ g,
                          const float* __restrict__ bta,
                          const float* __restrict__ mean,
                          const float* __restrict__ rstd,
                          unsigned short* __restrict__ out) {
  __shared__ float sh[C * SDIM];
  const int row = blockIdx.x;
  const int b   = row >> 10;
  const int rp  = row & 1023;
  const size_t rowbase = (size_t)b * NSPAT + (size_t)rp * SDIM;
  if (!XCL) {
    for (int e = threadIdx.x; e < C * SDIM; e += 256) {
      const int ci = e >> 5, xc = e & 31;
      sh[xc * C + ci] = x[((size_t)(b * C + ci)) * NSPAT + (size_t)rp * SDIM + xc];
    }
    __syncthreads();
  }
  for (int e = threadIdx.x; e < C * SDIM; e += 256) {
    const size_t gi = rowbase * C + e;
    const int ci = e & (C - 1);
    const float xv = XCL ? x[gi] : sh[e];
    const int bc = b * C + ci;
    const float nm = (xv - mean[bc]) * rstd[bc];
    float v = nm * (1.f + g[gi]) + bta[gi];
    v = (v > 0.f) ? v : 0.2f * v;
    out[gi] = f2bf(v);
  }
}

// ---------------------------------------------------------------------------
// Implicit-GEMM conv3d, bf16 WMMA, channels-last activations.
//   in  : bf16 [B][pos][CIN]
//   wt  : bf16 [cls][T][COUT][CIN]
//   One block = one (b,z,y) row (32 x positions) x all COUT.
//   blockDim = (COUT/16)*2 waves * 32.
// ---------------------------------------------------------------------------
template<int CIN, int COUT, bool K3, bool CLS, int ACT,
         bool OF32, bool OBF16, bool ADD, bool NCHW_OUT>
__global__ __launch_bounds__(COUT * 4)
void conv_wmma_kernel(const unsigned short* __restrict__ in,
                      const unsigned short* __restrict__ wt,
                      const float* __restrict__ bias,
                      const int* __restrict__ ycls,
                      float* __restrict__ out_f32,
                      unsigned short* __restrict__ out_bf16,
                      const float* __restrict__ addsrc) {
  constexpr int T      = K3 ? 27 : 1;
  constexpr int NLINES = K3 ? 9 : 1;
  constexpr int NTHR   = COUT * 4;
  __shared__ unsigned short smem[NLINES * 34 * CIN];

  const int row = blockIdx.x;
  const int b   = row >> 10;
  const int rp  = row & 1023;
  const int z   = rp >> 5;
  const int y   = rp & 31;
  const int tid = threadIdx.x;

  // ---- zero halo columns (xh = 0, 33) ----
  for (int i = tid; i < NLINES * CIN; i += NTHR) {
    const int line = i / CIN, ci = i & (CIN - 1);
    smem[(line * 34 + 0)  * CIN + ci] = 0;
    smem[(line * 34 + 33) * CIN + ci] = 0;
  }
  // ---- interior: contiguous b128 per halo line (async global->LDS on CDNA5) ----
  constexpr int LV = 32 * CIN / 8;           // 16B chunks per line
  for (int i = tid; i < NLINES * LV; i += NTHR) {
    const int line = i / LV;
    const int r    = i - line * LV;
    const int zz   = K3 ? (z + line / 3 - 1) : z;
    const int yy   = K3 ? (y + line % 3 - 1) : y;
    const bool valid = (zz >= 0) && (zz < SDIM) && (yy >= 0) && (yy < SDIM);
    uint4* dst = reinterpret_cast<uint4*>(&smem[(line * 34 + 1) * CIN]) + r;
    if (valid) {
      const uint4* src = reinterpret_cast<const uint4*>(
          in + ((size_t)b * NSPAT + ((size_t)(zz * SDIM + yy) * SDIM)) * CIN) + r;
#if defined(USE_ASYNC_LDS)
      __builtin_amdgcn_global_load_async_to_lds_b128(
          (as1_i32x4*)const_cast<uint4*>(src), (as3_i32x4*)dst, 0, 0);
#else
      *dst = *src;
#endif
    } else {
      *dst = make_uint4(0u, 0u, 0u, 0u);
    }
  }
#if defined(USE_ASYNC_LDS)
  wait_asynccnt0();   // each wave drains its own async copies before the barrier
#endif
  __syncthreads();

  // ---- per-wave 16x16 WMMA tile ----
  const int wave = tid >> 5;
  const int lane = tid & 31;
  constexpr int MT = COUT >> 4;
  const int mt  = wave % MT;
  const int nh  = wave / MT;
  const int lhi = lane >> 4;
  const int llo = lane & 15;
  const int co_base = mt * 16;
  const int xcol    = llo + nh * 16;

  const int cls = CLS ? ycls[b] : 0;
  const unsigned short* W = wt + (size_t)cls * (T * COUT * CIN);

  v8f acc = {};
  if (bias) {
    const float* bp = bias + cls * COUT;
    #pragma unroll
    for (int r = 0; r < 8; ++r) acc[r] = bp[co_base + r + lhi * 8];
  }

  #pragma unroll
  for (int tap = 0; tap < T; ++tap) {
    const int line = K3 ? ((tap / 9) * 3 + (tap / 3) % 3) : 0;
    const int xoff = K3 ? (tap % 3 - 1) : 0;
    const int xl   = xcol + xoff + 1;
    const unsigned short* wtap = W + ((size_t)tap * COUT + co_base) * CIN;
    const unsigned short* brow = &smem[(line * 34 + xl) * CIN];

    #pragma unroll
    for (int kc = 0; kc < CIN; kc += 32) {
      // A: m = llo; lanes 0-15 K 0-15, lanes 16-31 K 16-31
      const unsigned short* wa = wtap + llo * CIN + kc + lhi * 16;
      const v8bf a0 = *reinterpret_cast<const v8bf*>(wa);
      const v8bf a1 = *reinterpret_cast<const v8bf*>(wa + 8);
      const v16bf a = __builtin_shufflevector(a0, a1,
          0, 1, 2, 3, 4, 5, 6, 7, 8, 9, 10, 11, 12, 13, 14, 15);
      // B: n = llo; lanes 0-15 K 0-15, lanes 16-31 K 16-31 (32B contiguous)
      const v16bf bf = *reinterpret_cast<const v16bf*>(brow + kc + lhi * 16);
      acc = __builtin_amdgcn_wmma_f32_16x16x32_bf16(
          false, a, false, bf, (short)0, acc, false, false);
    }
  }

  // ---- epilogue ----
  const int pos = rp * SDIM + xcol;
  float v[8];
  const size_t obase = ((size_t)b * NSPAT + pos) * COUT + co_base + lhi * 8;
  #pragma unroll
  for (int r = 0; r < 8; ++r) {
    float t = acc[r];
    if (ADD) t += addsrc[obase + r];       // addsrc is channels-last
    if (ACT == 1) t = fmaxf(t, 0.f);
    v[r] = t;
  }
  if (NCHW_OUT) {
    #pragma unroll
    for (int r = 0; r < 8; ++r) {
      const int co = co_base + r + lhi * 8;
      out_f32[(size_t)(b * COUT + co) * NSPAT + pos] = v[r];
    }
  } else {
    if (OF32) {
      v4f lo = {v[0], v[1], v[2], v[3]};
      v4f hi = {v[4], v[5], v[6], v[7]};
      reinterpret_cast<v4f*>(out_f32 + obase)[0] = lo;
      reinterpret_cast<v4f*>(out_f32 + obase)[1] = hi;
    }
    if (OBF16) {
      ushort4 p0 = {f2bf(v[0]), f2bf(v[1]), f2bf(v[2]), f2bf(v[3])};
      ushort4 p1 = {f2bf(v[4]), f2bf(v[5]), f2bf(v[6]), f2bf(v[7])};
      reinterpret_cast<ushort4*>(out_bf16 + obase)[0] = p0;
      reinterpret_cast<ushort4*>(out_bf16 + obase)[1] = p1;
    }
  }
}

// ---------------------------------------------------------------------------
// Host-side orchestration
// ---------------------------------------------------------------------------
extern "C" void kernel_launch(void* const* d_in, const int* in_sizes, int n_in,
                              void* d_out, int out_size, void* d_ws, size_t ws_size,
                              hipStream_t stream) {
  (void)in_sizes; (void)n_in; (void)out_size; (void)ws_size;

  const float* x     = (const float*)d_in[0];
  const int*   ycls  = (const int*)d_in[1];
  const float* s0_ws = (const float*)d_in[2];
  const float* s0_bs = (const float*)d_in[3];
  const float* s0_wg = (const float*)d_in[4];
  const float* s0_bg = (const float*)d_in[5];
  const float* s0_wb = (const float*)d_in[6];
  const float* s0_bb = (const float*)d_in[7];
  const float* s1_ws = (const float*)d_in[8];
  const float* s1_bs = (const float*)d_in[9];
  const float* s1_wg = (const float*)d_in[10];
  const float* s1_bg = (const float*)d_in[11];
  const float* s1_wb = (const float*)d_in[12];
  const float* s1_bb = (const float*)d_in[13];
  const float* c0w   = (const float*)d_in[14];
  const float* c0b   = (const float*)d_in[15];
  const float* c1w   = (const float*)d_in[16];
  const float* c1b   = (const float*)d_in[17];
  const float* csw   = (const float*)d_in[18];

  char* ws = (char*)d_ws;
  size_t off = 0;
  auto alloc = [&](size_t bytes) -> char* {
    char* p = ws + off;
    off = (off + bytes + 255) & ~(size_t)255;
    return p;
  };

  // transposed bf16 weights [cls][T][Cout][Cin]
  unsigned short* wT_s0ws = (unsigned short*)alloc((size_t)4*27*64*64*2);
  unsigned short* wT_s0wg = (unsigned short*)alloc((size_t)4*27*64*64*2);
  unsigned short* wT_s0wb = (unsigned short*)alloc((size_t)4*27*64*64*2);
  unsigned short* wT_s1ws = (unsigned short*)alloc((size_t)4*27*64*32*2);
  unsigned short* wT_s1wg = (unsigned short*)alloc((size_t)4*27*32*64*2);
  unsigned short* wT_s1wb = (unsigned short*)alloc((size_t)4*27*32*64*2);
  unsigned short* wT_c0   = (unsigned short*)alloc((size_t)27*32*64*2);
  unsigned short* wT_c1   = (unsigned short*)alloc((size_t)27*32*32*2);
  unsigned short* wT_cs   = (unsigned short*)alloc((size_t)32*64*2);

  // channels-last activations
  unsigned short* x_cl  = (unsigned short*)alloc((size_t)NB*NSPAT*64*2);
  float* meanx = (float*)alloc(NB*64*4);
  float* rstdx = (float*)alloc(NB*64*4);
  float* xs    = (float*)alloc((size_t)NB*NSPAT*32*4);
  unsigned short* actv0 = (unsigned short*)alloc((size_t)NB*NSPAT*64*2);
  float* gamma0 = (float*)alloc((size_t)NB*NSPAT*64*4);
  float* beta0  = (float*)alloc((size_t)NB*NSPAT*64*4);
  unsigned short* t0    = (unsigned short*)alloc((size_t)NB*NSPAT*64*2);
  float* dx     = (float*)alloc((size_t)NB*NSPAT*32*4);
  unsigned short* dx_bf = (unsigned short*)alloc((size_t)NB*NSPAT*32*2);
  float* meand = (float*)alloc(NB*32*4);
  float* rstdd = (float*)alloc(NB*32*4);
  unsigned short* actv1 = (unsigned short*)alloc((size_t)NB*NSPAT*64*2);
  float* gamma1 = (float*)alloc((size_t)NB*NSPAT*32*4);
  float* beta1  = (float*)alloc((size_t)NB*NSPAT*32*4);
  unsigned short* t1    = (unsigned short*)alloc((size_t)NB*NSPAT*32*2);

  auto wtrans = [&](const float* w, unsigned short* o, int Cout, int Cin, int T, int NC) {
    int total = NC * Cout * Cin * T;
    wtrans_kernel<<<(total + 255) / 256, 256, 0, stream>>>(w, o, Cout, Cin, T, total);
  };
  wtrans(s0_ws, wT_s0ws, 64, 64, 27, 4);
  wtrans(s0_wg, wT_s0wg, 64, 64, 27, 4);
  wtrans(s0_wb, wT_s0wb, 64, 64, 27, 4);
  wtrans(s1_ws, wT_s1ws, 64, 32, 27, 4);
  wtrans(s1_wg, wT_s1wg, 32, 64, 27, 4);
  wtrans(s1_wb, wT_s1wb, 32, 64, 27, 4);
  wtrans(c0w,   wT_c0,   32, 64, 27, 1);
  wtrans(c1w,   wT_c1,   32, 32, 27, 1);
  wtrans(csw,   wT_cs,   32, 64, 1, 1);

  const int ROWS = NB * SDIM * SDIM;  // 4096

  cvt_cl_kernel<64><<<ROWS, 256, 0, stream>>>(x, x_cl);
  stats_kernel<<<NB * 64, 256, 0, stream>>>(x, meanx, rstdx);

  // shortcut: 1x1x1, 64->32, no bias, f32 CL out
  conv_wmma_kernel<64,32,false,false,0,true,false,false,false>
      <<<ROWS, 128, 0, stream>>>(x_cl, wT_cs, nullptr, nullptr, xs, nullptr, nullptr);

  // SPADE0 actv = relu(conv3(x)), 64->64, bf16 CL out
  conv_wmma_kernel<64,64,true,true,1,false,true,false,false>
      <<<ROWS, 256, 0, stream>>>(x_cl, wT_s0ws, s0_bs, ycls, nullptr, actv0, nullptr);
  // gamma0 / beta0: 64->64, f32 CL out
  conv_wmma_kernel<64,64,true,true,0,true,false,false,false>
      <<<ROWS, 256, 0, stream>>>(actv0, wT_s0wg, s0_bg, ycls, gamma0, nullptr, nullptr);
  conv_wmma_kernel<64,64,true,true,0,true,false,false,false>
      <<<ROWS, 256, 0, stream>>>(actv0, wT_s0wb, s0_bb, ycls, beta0, nullptr, nullptr);
  spade_combine_kernel<64,false><<<ROWS, 256, 0, stream>>>(
      x, gamma0, beta0, meanx, rstdx, t0);

  // conv0: 64->32, dual f32+bf16 CL out (raw, no act)
  conv_wmma_kernel<64,32,true,false,0,true,true,false,false>
      <<<ROWS, 128, 0, stream>>>(t0, wT_c0, c0b, nullptr, dx, dx_bf, nullptr);
  stats_cl_kernel<32><<<NB, 256, 0, stream>>>(dx, meand, rstdd);

  // SPADE1 actv = relu(conv3(dx)), 32->64
  conv_wmma_kernel<32,64,true,true,1,false,true,false,false>
      <<<ROWS, 256, 0, stream>>>(dx_bf, wT_s1ws, s1_bs, ycls, nullptr, actv1, nullptr);
  // gamma1 / beta1: 64->32
  conv_wmma_kernel<64,32,true,true,0,true,false,false,false>
      <<<ROWS, 128, 0, stream>>>(actv1, wT_s1wg, s1_bg, ycls, gamma1, nullptr, nullptr);
  conv_wmma_kernel<64,32,true,true,0,true,false,false,false>
      <<<ROWS, 128, 0, stream>>>(actv1, wT_s1wb, s1_bb, ycls, beta1, nullptr, nullptr);
  spade_combine_kernel<32,true><<<ROWS, 256, 0, stream>>>(
      dx, gamma1, beta1, meand, rstdd, t1);

  // conv1: 32->32 + residual (xs, CL) -> final NCDHW f32 output
  conv_wmma_kernel<32,32,true,false,0,true,false,true,true>
      <<<ROWS, 128, 0, stream>>>(t1, wT_c1, c1b, nullptr,
                                 (float*)d_out, nullptr, xs);
}
